// PointerDecoder_86234353369880
// MI455X (gfx1250) — matrix operand order repaired and checked by reading, not compiled, not run on hardware
//
#include <hip/hip_runtime.h>

// ---------------------------------------------------------------------------
// PointerDecoder for MI455X (gfx1250, wave32).
// Phase 1: K16 = f16(node_embeds @ Wk^T) via v_wmma_f32_16x16x32_f16.
// Phase 2: persistent 64-block decoder, global barrier via atomics,
//          Threefry-2x32 Gumbel-max categorical sampling, GRU cell.
// Workspace: [0, 4MB) = K16 (8192*256 f16), then DecState.
// ---------------------------------------------------------------------------

#define N_NODES 8192
#define DIM 256
#define NBLK 64          // persistent decode blocks (8192/64 = 128 rows each)
#define NTHR 256
#define NEG_BIG -1.0e9f
#define LOG_CLAMP -20.72326583694641f  // log(1e-9)

typedef __attribute__((ext_vector_type(16))) _Float16 v16h;
typedef __attribute__((ext_vector_type(8)))  _Float16 v8h;
typedef __attribute__((ext_vector_type(8)))  float    v8f;

struct DecState {
  float    hidden[DIM];
  _Float16 q16[DIM];
  float    pmax[NBLK];
  float    pbestv[NBLK];
  int      pbesti[NBLK];
  float    psum[NBLK];
  float    gmax;
  int      nxt;
  float    masked_nxt;
  int      current;
  int      done;
  float    logp;
  unsigned gen;
  unsigned cnt;
};

// ---------------- Threefry-2x32-20 (JAX-compatible schedule) ----------------
__device__ __forceinline__ uint2 threefry2x32(unsigned k0, unsigned k1,
                                              unsigned c0, unsigned c1) {
  unsigned ks0 = k0, ks1 = k1, ks2 = 0x1BD11BDAu ^ k0 ^ k1;
  unsigned x0 = c0 + ks0, x1 = c1 + ks1;
#define TF_ROT(x, r) (((x) << (r)) | ((x) >> (32 - (r))))
#define TF_R(r) { x0 += x1; x1 = TF_ROT(x1, r); x1 ^= x0; }
  TF_R(13) TF_R(15) TF_R(26) TF_R(6)   x0 += ks1; x1 += ks2 + 1u;
  TF_R(17) TF_R(29) TF_R(16) TF_R(24)  x0 += ks2; x1 += ks0 + 2u;
  TF_R(13) TF_R(15) TF_R(26) TF_R(6)   x0 += ks0; x1 += ks1 + 3u;
  TF_R(17) TF_R(29) TF_R(16) TF_R(24)  x0 += ks1; x1 += ks2 + 4u;
  TF_R(13) TF_R(15) TF_R(26) TF_R(6)   x0 += ks2; x1 += ks0 + 5u;
#undef TF_R
#undef TF_ROT
  uint2 r; r.x = x0; r.y = x1; return r;
}

// keys = split(key(1), 64): counts 0..127 split into halves (x0=j, x1=64+j)
__device__ __forceinline__ uint2 jax_stepkey(int i, int nsteps) {
  unsigned j0 = 2u * (unsigned)i, j1 = j0 + 1u;
  unsigned n = (unsigned)nsteps;
  unsigned w0 = (j0 < n) ? threefry2x32(0u, 1u, j0, n + j0).x
                         : threefry2x32(0u, 1u, j0 - n, j0).y;
  unsigned w1 = (j1 < n) ? threefry2x32(0u, 1u, j1, n + j1).x
                         : threefry2x32(0u, 1u, j1 - n, j1).y;
  uint2 k; k.x = w0; k.y = w1; return k;
}

// Gumbel noise for logit n of an 8192-vector under step key sk.
__device__ __forceinline__ float gumbel_for(uint2 sk, int n) {
  const int half = N_NODES / 2;
  uint2 r = (n < half) ? threefry2x32(sk.x, sk.y, (unsigned)n, (unsigned)(half + n))
                       : threefry2x32(sk.x, sk.y, (unsigned)(n - half), (unsigned)n);
  unsigned bits = (n < half) ? r.x : r.y;
  float u = __uint_as_float((bits >> 9) | 0x3f800000u) - 1.0f;  // [0,1)
  u = fmaxf(u, 1.1754944e-38f);
  return -logf(-logf(u));
}

__device__ __forceinline__ float sigmoidf(float x) { return 1.0f / (1.0f + expf(-x)); }

// ---------------- Grid-wide barrier (generation counter) ----------------
__device__ __forceinline__ void gsync(DecState* S, unsigned& lgen) {
  __threadfence();
  __syncthreads();
  if (threadIdx.x == 0) {
    if (atomicAdd(&S->cnt, 1u) == NBLK - 1) {
      atomicExch(&S->cnt, 0u);
      __threadfence();
      atomicAdd(&S->gen, 1u);
    } else {
      while (atomicAdd(&S->gen, 0u) <= lgen) {
        __builtin_amdgcn_s_sleep(2);
      }
    }
  }
  lgen++;
  __syncthreads();
  __threadfence();
}

// ---------------- Phase 0: state init (every launch, deterministic) --------
__global__ void pd_init(const float* __restrict__ node_embeds,
                        const int* __restrict__ start_p, DecState* S) {
  int t = threadIdx.x;
  int start = *start_p;
  S->hidden[t] = node_embeds[(size_t)start * DIM + t];
  if (t == 0) {
    S->current = start;
    S->done = 0;
    S->logp = 0.0f;
    S->gen = 0u;
    S->cnt = 0u;
    S->gmax = 0.0f;
    S->nxt = 0;
    S->masked_nxt = 0.0f;
  }
}

// ---------------- Phase 1: K16 = f16(node_embeds @ Wk^T) via WMMA ----------
// Grid: 512 blocks (one 16-row strip), 256 threads = 8 waves.
// Wave w computes output tiles at column-tiles {2w, 2w+1}; K unrolled 8x32.
__global__ void pd_gemm_k(const float* __restrict__ A,   // node_embeds [8192,256]
                          const float* __restrict__ Wk,  // [256,256]
                          _Float16* __restrict__ K16) {  // [8192,256]
  const int strip = blockIdx.x;               // M tile: rows strip*16..+15
  const int lane  = threadIdx.x & 31;
  const int wave  = threadIdx.x >> 5;
  const int mrow  = lane & 15;
  const bool hi   = (lane & 16) != 0;
  const int r     = strip * 16 + mrow;        // A row this lane supplies

#pragma unroll
  for (int tn = 0; tn < 2; ++tn) {
    const int nt = wave * 2 + tn;             // column tile 0..15
    const int d  = nt * 16 + mrow;            // Wk row == output column
    v8f c = {};
#pragma unroll
    for (int kc = 0; kc < 8; ++kc) {
      // ---- A fragment: 16x32, k contiguous per lane half (same order as B) ----
      // lanes 0-15: K = kc*32 + 0..15; lanes 16-31: K = kc*32 + 16..31
      const int kb = kc * 32 + (hi ? 16 : 0);
      const float4* ap0 = reinterpret_cast<const float4*>(A + (size_t)r * DIM + kb);
      const float4* ap1 = reinterpret_cast<const float4*>(A + (size_t)r * DIM + kb + 8);
      float ta[16];
      *reinterpret_cast<float4*>(&ta[0])  = ap0[0];
      *reinterpret_cast<float4*>(&ta[4])  = ap0[1];
      *reinterpret_cast<float4*>(&ta[8])  = ap1[0];
      *reinterpret_cast<float4*>(&ta[12]) = ap1[1];
      v16h a;
#pragma unroll
      for (int e = 0; e < 16; ++e) a[e] = (_Float16)ta[e];

      // ---- B fragment: 32x16, B[k][col] = Wk[col][k] (contiguous in k) ----
      // lanes 0-15: K = kc*32 + 0..15; lanes 16-31: K = kc*32 + 16..31
      const int ks = kc * 32 + (hi ? 16 : 0);
      const float4* bp = reinterpret_cast<const float4*>(Wk + (size_t)d * DIM + ks);
      float tb[16];
      *reinterpret_cast<float4*>(&tb[0])  = bp[0];
      *reinterpret_cast<float4*>(&tb[4])  = bp[1];
      *reinterpret_cast<float4*>(&tb[8])  = bp[2];
      *reinterpret_cast<float4*>(&tb[12]) = bp[3];
      v16h b;
#pragma unroll
      for (int e = 0; e < 16; ++e) b[e] = (_Float16)tb[e];

      c = __builtin_amdgcn_wmma_f32_16x16x32_f16(
          /*neg_a=*/false, a, /*neg_b=*/false, b,
          /*c_mod=*/(short)0, c, /*reuse_a=*/false, /*reuse_b=*/false);
    }
    // C/D layout: VGPR v -> row v + (hi?8:0), col = lane&15 within tile
#pragma unroll
    for (int v = 0; v < 8; ++v) {
      const int row = strip * 16 + v + (hi ? 8 : 0);
      K16[(size_t)row * DIM + nt * 16 + mrow] = (_Float16)c[v];
    }
  }
}

// ---------------- Phase 2: persistent sequential decoder -------------------
__global__ void pd_decode(const float* __restrict__ node_embeds,
                          const float* __restrict__ adj_w,
                          const float* __restrict__ Wq,
                          const float* __restrict__ W_ih,
                          const float* __restrict__ W_hh,
                          const float* __restrict__ b_ih,
                          const float* __restrict__ b_hh,
                          const _Float16* __restrict__ K16,
                          DecState* S,
                          const int* __restrict__ target_p,
                          const int* __restrict__ max_steps_p,
                          float* __restrict__ out) {
  __shared__ float q_lds[DIM];
  __shared__ float h_lds[DIM];
  __shared__ float x_lds[DIM];
  __shared__ float sc_lds[N_NODES / NBLK];      // 128 masked scores (this block)
  __shared__ float red_max[8], red_bv[8];
  __shared__ int   red_bi[8];
  __shared__ float red_sum[NTHR];
  __shared__ int   ibcast[2];

  const int b = blockIdx.x, t = threadIdx.x;
  const int lane = t & 31, wave = t >> 5;
  const int rows_per_blk = N_NODES / NBLK;       // 128
  const int target = *target_p;
  const int msteps = *max_steps_p;
  unsigned lgen = 0;

  for (int step = 0; step < msteps; ++step) {
    // ---- block 0: q16 = f16((Wq @ hidden) * 1/sqrt(D)) ----
    if (b == 0) {
      h_lds[t] = S->hidden[t];
      __syncthreads();
      float acc = 0.0f;
      const float* wr = Wq + (size_t)t * DIM;
#pragma unroll 8
      for (int k = 0; k < DIM; ++k) acc += wr[k] * h_lds[k];
      S->q16[t] = (_Float16)(acc * 0.0625f);     // fold scale 1/16
    }
    gsync(S, lgen);

    // ---- all blocks: masked scores + Gumbel argmax over own 128 rows ----
    q_lds[t] = (float)S->q16[t];
    __syncthreads();
    const int cur = S->current;
    const uint2 sk = jax_stepkey(step, msteps);

    float wmax = -3.402823466e38f, wbv = -3.402823466e38f;
    int wbi = -1;
#pragma unroll 1
    for (int i = 0; i < 16; ++i) {
      const int row = b * rows_per_blk + wave * 16 + i;
      const _Float16* kp = K16 + (size_t)row * DIM + lane * 8;
      __builtin_prefetch(kp + DIM, 0, 1);        // global_prefetch_b8
      v8h kv = *reinterpret_cast<const v8h*>(kp);
      float acc = 0.0f;
#pragma unroll
      for (int e = 0; e < 8; ++e) acc += (float)kv[e] * q_lds[lane * 8 + e];
#pragma unroll
      for (int off = 16; off > 0; off >>= 1) acc += __shfl_xor(acc, off, 32);
      if (lane == 0) {
        const float aw = adj_w[(size_t)cur * N_NODES + row];
        const bool m = (aw > 0.0f) && (row != cur);
        const float masked = m ? acc : NEG_BIG;
        sc_lds[wave * 16 + i] = masked;
        wmax = fmaxf(wmax, masked);
        const float pert = masked + gumbel_for(sk, row);
        if (pert > wbv) { wbv = pert; wbi = row; }
      }
    }
    if (lane == 0) { red_max[wave] = wmax; red_bv[wave] = wbv; red_bi[wave] = wbi; }
    __syncthreads();
    if (t == 0) {
      float m = -3.402823466e38f, bv = -3.402823466e38f; int bi = -1;
      for (int w = 0; w < 8; ++w) {
        m = fmaxf(m, red_max[w]);
        if (red_bv[w] > bv) { bv = red_bv[w]; bi = red_bi[w]; }
      }
      S->pmax[b] = m; S->pbestv[b] = bv; S->pbesti[b] = bi;
    }
    gsync(S, lgen);

    // ---- block 0: global max + sampled index (deterministic order) ----
    if (b == 0 && t == 0) {
      float m = -3.402823466e38f, bv = -3.402823466e38f; int bi = -1;
      for (int i = 0; i < NBLK; ++i) {
        m = fmaxf(m, S->pmax[i]);
        if (S->pbestv[i] > bv) { bv = S->pbestv[i]; bi = S->pbesti[i]; }
      }
      S->gmax = m; S->nxt = bi;
    }
    gsync(S, lgen);

    // ---- all blocks: partial sum-exp (ordered reduction -> deterministic) --
    const float gm = S->gmax;
    const int nxt = S->nxt;
    red_sum[t] = (t < rows_per_blk) ? expf(sc_lds[t] - gm) : 0.0f;
    __syncthreads();
    for (int s2 = NTHR / 2; s2 > 0; s2 >>= 1) {
      if (t < s2) red_sum[t] += red_sum[t + s2];
      __syncthreads();
    }
    if (t == 0) {
      S->psum[b] = red_sum[0];
      if (nxt >= b * rows_per_blk && nxt < (b + 1) * rows_per_blk)
        S->masked_nxt = sc_lds[nxt - b * rows_per_blk];
    }
    gsync(S, lgen);

    // ---- block 0: finalize step: logp/done/current/path + GRU update ----
    if (b == 0) {
      if (t == 0) {
        float ssum = 0.0f;
        for (int i = 0; i < NBLK; ++i) ssum += S->psum[i];
        const float lse = gm + logf(ssum);
        const float lp = fmaxf(S->masked_nxt - lse, LOG_CLAMP);
        const bool no_valid = (gm <= -1.0e8f);
        const bool was_done = (S->done != 0);
        const bool active = !(was_done || no_valid);
        out[step] = active ? (float)nxt : -1.0f;
        if (active) { S->logp += lp; S->current = nxt; }
        S->done = (was_done || no_valid || (active && nxt == target)) ? 1 : 0;
        ibcast[0] = active ? 1 : 0;
      }
      __syncthreads();
      if (ibcast[0]) {
        // GRU cell: h' = (1-z)*n + z*h
        x_lds[t] = node_embeds[(size_t)S->nxt * DIM + t];
        h_lds[t] = S->hidden[t];
        __syncthreads();
        const float* wir = W_ih + (size_t)t * DIM;
        const float* wiz = W_ih + (size_t)(DIM + t) * DIM;
        const float* win = W_ih + (size_t)(2 * DIM + t) * DIM;
        const float* whr = W_hh + (size_t)t * DIM;
        const float* whz = W_hh + (size_t)(DIM + t) * DIM;
        const float* whn = W_hh + (size_t)(2 * DIM + t) * DIM;
        float gr = b_ih[t], gz = b_ih[DIM + t], gn = b_ih[2 * DIM + t];
        float hr = b_hh[t], hz = b_hh[DIM + t], hn = b_hh[2 * DIM + t];
#pragma unroll 4
        for (int k = 0; k < DIM; ++k) {
          const float xv = x_lds[k], hv = h_lds[k];
          gr += wir[k] * xv; gz += wiz[k] * xv; gn += win[k] * xv;
          hr += whr[k] * hv; hz += whz[k] * hv; hn += whn[k] * hv;
        }
        const float rg = sigmoidf(gr + hr);
        const float zg = sigmoidf(gz + hz);
        const float ng = tanhf(gn + rg * hn);
        S->hidden[t] = (1.0f - zg) * ng + zg * h_lds[t];
      }
    }
    gsync(S, lgen);
  }
  if (b == 0 && t == 0) out[msteps] = S->logp;
}

// ---------------------------------------------------------------------------
extern "C" void kernel_launch(void* const* d_in, const int* in_sizes, int n_in,
                              void* d_out, int out_size, void* d_ws, size_t ws_size,
                              hipStream_t stream) {
  const float* node_embeds = (const float*)d_in[0];
  const float* adj_w       = (const float*)d_in[1];
  const float* Wq          = (const float*)d_in[2];
  const float* Wk          = (const float*)d_in[3];
  const float* W_ih        = (const float*)d_in[4];
  const float* W_hh        = (const float*)d_in[5];
  const float* b_ih        = (const float*)d_in[6];
  const float* b_hh        = (const float*)d_in[7];
  const int* start_p       = (const int*)d_in[8];
  const int* target_p      = (const int*)d_in[9];
  const int* max_steps_p   = (const int*)d_in[10];

  // workspace: K16 (4MB) then DecState. Requires ws_size >= 4MB + ~3KB.
  _Float16* K16 = (_Float16*)d_ws;
  DecState* S = (DecState*)((char*)d_ws + (size_t)N_NODES * DIM * sizeof(_Float16));
  float* out = (float*)d_out;

  pd_init<<<dim3(1), dim3(NTHR), 0, stream>>>(node_embeds, start_p, S);
  pd_gemm_k<<<dim3(N_NODES / 16), dim3(NTHR), 0, stream>>>(node_embeds, Wk, K16);
  pd_decode<<<dim3(NBLK), dim3(NTHR), 0, stream>>>(
      node_embeds, adj_w, Wq, W_ih, W_hh, b_ih, b_hh, K16, S,
      target_p, max_steps_p, out);
}
